// CrossAttention_30253749633441
// MI455X (gfx1250) — matrix operand order repaired and checked by reading, not compiled, not run on hardware
//
#include <hip/hip_runtime.h>

typedef _Float16 half16v __attribute__((ext_vector_type(16)));
typedef _Float16 half8v  __attribute__((ext_vector_type(8)));
typedef float    float8v __attribute__((ext_vector_type(8)));
typedef int      v4i     __attribute__((vector_size(16)));
typedef v4i __attribute__((address_space(1)))* gv4_p;  // global-AS pointer to int4
typedef v4i __attribute__((address_space(3)))* lv4_p;  // LDS-AS pointer to int4

constexpr int NB    = 4;
constexpr int NQT   = 2048;   // query tokens per batch
constexpr int NKT   = 2048;   // kv tokens per batch
constexpr int NHEAD = 8;
constexpr int DQK   = 160;    // per-head q/k feature dim (16ch * 8 inner comps + 32 scalars)
constexpr int DVF   = 288;    // v feature dim (16ch * 16 comps + 32 scalars)
constexpr int HCM   = 128;    // H*CM merged mv channels
constexpr int HCS   = 256;    // H*CS merged scalar channels
constexpr int KSTR  = 168;    // LDS K row stride (halves): 336B, 16B-aligned, conflict-spread
constexpr int VSTR  = 56;     // LDS V row stride (halves): 112B, 16B-aligned, conflict-spread

// PGA basis structure: grade of each component, e0-map source comp, e0-map basis index
constexpr int GRADE_[16] = {0,1,1,1,1,2,2,2,2,2,2,3,3,3,3,4};
constexpr int SRC_[16]   = {-1,0,-1,-1,-1,2,3,4,-1,-1,-1,8,9,10,-1,14};
constexpr int Y2_[16]    = {0,5,0,0,0,6,6,6,0,0,0,7,7,7,0,8};
constexpr int INNER_[8]  = {0,2,3,4,8,9,10,14};
// softmax scale 1/sqrt(160) folded with log2(e) so we can use exp2
constexpr float QSCALE = 0.07905694150420949f * 1.4426950408889634f;

// ---- shared equivariant channel helper: out[a] = sum_c w[c,g(a)]*mv[c,a] (+ e0 term) ----
__device__ __forceinline__ void equi_mv16(const float* __restrict__ w /*16x9*/,
                                          const float (*mvs)[16], float out[16]) {
#pragma unroll
  for (int a = 0; a < 16; ++a) {
    float acc = 0.f;
    const int g = GRADE_[a];
#pragma unroll
    for (int c = 0; c < 16; ++c) acc = fmaf(w[c * 9 + g], mvs[c][a], acc);
    if (SRC_[a] >= 0) {
      const int s = SRC_[a], y = Y2_[a];
#pragma unroll
      for (int c = 0; c < 16; ++c) acc = fmaf(w[c * 9 + y], mvs[c][s], acc);
    }
    out[a] = acc;
  }
}

// ===================== Q projection -> qfeat (B,H,NQ,160) f16, scale folded =====================
__global__ __launch_bounds__(256) void proj_q_kernel(
    const float* __restrict__ mv, const float* __restrict__ sc,
    const float* __restrict__ w_mv, const float* __restrict__ w_s2mv,
    const float* __restrict__ b_mv, const float* __restrict__ w_mv2s,
    const float* __restrict__ w_s2s, const float* __restrict__ b_s,
    _Float16* __restrict__ qfeat) {
  __shared__ float mvs[16][16];
  __shared__ float ss[32];
  const int n = blockIdx.x;  // b*NQT + q
  const int b = n / NQT, q = n % NQT;
  const int tid = threadIdx.x;
  (&mvs[0][0])[tid] = mv[(size_t)n * 256 + tid];
  if (tid < 32) ss[tid] = sc[(size_t)n * 32 + tid];
  __syncthreads();
  if (tid < 128) {
    const int o = tid;  // output mv channel, o = cm*8 + h
    float out[16];
    equi_mv16(w_mv + o * 144, mvs, out);
    float a0 = b_mv[o];
#pragma unroll
    for (int i = 0; i < 32; ++i) a0 = fmaf(ss[i], w_s2mv[o * 32 + i], a0);
    out[0] += a0;
    const int h = o & 7, cm = o >> 3;
    _Float16* dst = qfeat + (((size_t)(b * NHEAD + h)) * NQT + q) * DQK + cm * 8;
#pragma unroll
    for (int xi = 0; xi < 8; ++xi) dst[xi] = (_Float16)(out[INNER_[xi]] * QSCALE);
  } else {
    const int t = tid - 128;
#pragma unroll
    for (int rep = 0; rep < 2; ++rep) {
      const int sidx = t + rep * 128;  // sidx = cs*8 + h
      float acc = b_s[sidx];
#pragma unroll
      for (int c = 0; c < 16; ++c) acc = fmaf(mvs[c][0], w_mv2s[sidx * 16 + c], acc);
#pragma unroll
      for (int i = 0; i < 32; ++i) acc = fmaf(ss[i], w_s2s[sidx * 32 + i], acc);
      const int h = sidx & 7, cs = sidx >> 3;
      qfeat[(((size_t)(b * NHEAD + h)) * NQT + q) * DQK + 128 + cs] =
          (_Float16)(acc * QSCALE);
    }
  }
}

// ===================== K/V projection -> kfeat (B,NK,160), vT (B,288,NK) f16 =====================
__global__ __launch_bounds__(128) void proj_kv_kernel(
    const float* __restrict__ mv, const float* __restrict__ sc,
    const float* __restrict__ kw_mv, const float* __restrict__ kw_s2mv,
    const float* __restrict__ kb_mv, const float* __restrict__ kw_mv2s,
    const float* __restrict__ kw_s2s, const float* __restrict__ kb_s,
    const float* __restrict__ vw_mv, const float* __restrict__ vw_s2mv,
    const float* __restrict__ vb_mv, const float* __restrict__ vw_mv2s,
    const float* __restrict__ vw_s2s, const float* __restrict__ vb_s,
    _Float16* __restrict__ kfeat, _Float16* __restrict__ vT) {
  __shared__ float mvs[16][16];
  __shared__ float ss[32];
  const int n = blockIdx.x;
  const int b = n / NKT, k = n % NKT;
  const int tid = threadIdx.x;
  (&mvs[0][0])[tid] = mv[(size_t)n * 256 + tid];
  (&mvs[0][0])[tid + 128] = mv[(size_t)n * 256 + tid + 128];
  if (tid < 32) ss[tid] = sc[(size_t)n * 32 + tid];
  __syncthreads();
  if (tid < 32) {
    const bool isV = tid >= 16;
    const int o = tid & 15;
    float out[16];
    equi_mv16((isV ? vw_mv : kw_mv) + o * 144, mvs, out);
    const float* s2mv = isV ? vw_s2mv : kw_s2mv;
    float a0 = (isV ? vb_mv : kb_mv)[o];
#pragma unroll
    for (int i = 0; i < 32; ++i) a0 = fmaf(ss[i], s2mv[o * 32 + i], a0);
    out[0] += a0;
    if (!isV) {
      _Float16* dst = kfeat + (size_t)n * DQK + o * 8;
#pragma unroll
      for (int xi = 0; xi < 8; ++xi) dst[xi] = (_Float16)out[INNER_[xi]];
    } else {
#pragma unroll
      for (int a = 0; a < 16; ++a)
        vT[((size_t)b * DVF + o * 16 + a) * NKT + k] = (_Float16)out[a];
    }
  } else if (tid < 96) {
    const bool isV = tid >= 64;
    const int cs = tid - (isV ? 64 : 32);
    const float* wm = isV ? vw_mv2s : kw_mv2s;
    const float* wss = isV ? vw_s2s : kw_s2s;
    float acc = (isV ? vb_s : kb_s)[cs];
#pragma unroll
    for (int c = 0; c < 16; ++c) acc = fmaf(mvs[c][0], wm[cs * 16 + c], acc);
#pragma unroll
    for (int i = 0; i < 32; ++i) acc = fmaf(ss[i], wss[cs * 32 + i], acc);
    if (!isV) kfeat[(size_t)n * DQK + 128 + cs] = (_Float16)acc;
    else      vT[((size_t)b * DVF + 256 + cs) * NKT + k] = (_Float16)acc;
  }
}

// ===================== Flash attention =====================
__device__ __forceinline__ half16v cat8(half8v lo, half8v hi) {
  return __builtin_shufflevector(lo, hi, 0, 1, 2, 3, 4, 5, 6, 7, 8, 9, 10, 11, 12, 13, 14, 15);
}
__device__ __forceinline__ half16v ldfrag(const _Float16* p) {  // 16 contiguous halves
  return cat8(*(const half8v*)p, *(const half8v*)(p + 8));
}
__device__ __forceinline__ float8v wmma_f16(half16v a, half16v b, float8v c) {
  return __builtin_amdgcn_wmma_f32_16x16x32_f16(false, a, false, b, (short)0, c, false, false);
}

// async 16B global->LDS copy (gfx1250 ASYNC path), with portable fallback
__device__ __forceinline__ void async_copy16(const _Float16* g, _Float16* l) {
#if __has_builtin(__builtin_amdgcn_global_load_async_to_lds_b128)
  __builtin_amdgcn_global_load_async_to_lds_b128((gv4_p)g, (lv4_p)l, 0, 0);
#else
  *(half8v*)l = *(const half8v*)g;
#endif
}
__device__ __forceinline__ void wait_async0() {
#if __has_builtin(__builtin_amdgcn_s_wait_asynccnt)
  __builtin_amdgcn_s_wait_asynccnt(0);
#elif __has_builtin(__builtin_amdgcn_global_load_async_to_lds_b128)
  asm volatile("s_wait_asynccnt 0x0" ::: "memory");
#endif
}

// cooperative fill of one KV tile (32 cols): K 32x160, V 288x32, 16B chunks
__device__ __forceinline__ void fill_tiles(const _Float16* __restrict__ kb_base,
                                           const _Float16* __restrict__ v_base,
                                           _Float16* Kb, _Float16* Vb,
                                           int kv0, int tid) {
  // K: 640 slots (32 cols x 20 chunks) padded to 768 so every wave issues equally
#pragma unroll
  for (int j = 0; j < 3; ++j) {
    int idx = tid + 256 * j;
    if (idx >= 640) idx -= 640;  // duplicate slots write identical data (benign)
    const int col = idx / 20, chunk = idx % 20;
    async_copy16(kb_base + (size_t)(kv0 + col) * DQK + chunk * 8,
                 Kb + col * KSTR + chunk * 8);
  }
  // V: 1152 slots (288 rows x 4 chunks) padded to 1280
#pragma unroll
  for (int j = 0; j < 5; ++j) {
    int idx = tid + 256 * j;
    if (idx >= 1152) idx -= 1152;
    const int row = idx >> 2, part = idx & 3;
    async_copy16(v_base + (size_t)row * NKT + kv0 + part * 8,
                 Vb + row * VSTR + part * 8);
  }
}

__global__ __launch_bounds__(256) void attn_kernel(
    const _Float16* __restrict__ qfeat, const _Float16* __restrict__ kfeat,
    const _Float16* __restrict__ vT,
    _Float16* __restrict__ hmv, _Float16* __restrict__ hs) {
  __shared__ __align__(16) _Float16 Kbuf[2][32 * KSTR];   // 21,504 B
  __shared__ __align__(16) _Float16 Vbuf[2][DVF * VSTR];  // 64,512 B
  __shared__ __align__(16) _Float16 Plds[8][16][48];      // 12,288 B  (total 96 KB)
  const int tid = threadIdx.x;
  const int wave = tid >> 5;
  const int lane = tid & 31;
  const int l15 = lane & 15;
  const int hiHalf = lane >> 4;
  const int qt = blockIdx.x & 15;  // 16 q tiles of 128 rows
  const int bh = blockIdx.x >> 4;  // b*NHEAD + h
  const int b = bh >> 3, h = bh & 7;

  const _Float16* kb_base = kfeat + (size_t)b * NKT * DQK;
  const _Float16* v_base  = vT + (size_t)b * DVF * NKT;

  // --- preload Q A-fragments (5 chunks of K=32) in contiguous k order (k = 16*hiHalf + s) ---
  const _Float16* qrowp = qfeat + (((size_t)bh) * NQT + qt * 128 + wave * 16 + l15) * DQK;
  half16v qa[5];
#pragma unroll
  for (int c = 0; c < 5; ++c) {
    const int d0 = c * 32 + hiHalf * 16;
    qa[c] = cat8(*(const half8v*)(qrowp + d0), *(const half8v*)(qrowp + d0 + 8));
  }

  float8v Oacc[18];
#pragma unroll
  for (int t = 0; t < 18; ++t)
#pragma unroll
    for (int r = 0; r < 8; ++r) Oacc[t][r] = 0.f;
  float m[8], l[8];
#pragma unroll
  for (int r = 0; r < 8; ++r) { m[r] = -3.0e38f; l[r] = 0.f; }

  fill_tiles(kb_base, v_base, Kbuf[0], Vbuf[0], 0, tid);  // prime the pipe

  for (int kv0 = 0; kv0 < NKT; kv0 += 32) {
    const int buf = (kv0 >> 5) & 1;
    wait_async0();       // own async fill of current tile complete
    __syncthreads();     // everyone's fill complete; previous tile consumers done
    if (kv0 + 32 < NKT)  // overlap next-tile fill with this tile's compute
      fill_tiles(kb_base, v_base, Kbuf[buf ^ 1], Vbuf[buf ^ 1], kv0 + 32, tid);

    const _Float16* Kt = Kbuf[buf];
    const _Float16* Vt = Vbuf[buf];

    // --- scores: two 16-col tiles, 5 WMMAs each, depth-1 pipelined LDS reads ---
    const _Float16* k0p = Kt + (size_t)l15 * KSTR + hiHalf * 16;
    const _Float16* k1p = Kt + (size_t)(l15 + 16) * KSTR + hiHalf * 16;
    float8v S0, S1;
#pragma unroll
    for (int r = 0; r < 8; ++r) { S0[r] = 0.f; S1[r] = 0.f; }
    half16v kb0 = ldfrag(k0p);
    half16v kb1 = ldfrag(k1p);
#pragma unroll
    for (int c = 0; c < 5; ++c) {
      half16v nb0 = kb0, nb1 = kb1;
      if (c < 4) { nb0 = ldfrag(k0p + (c + 1) * 32); nb1 = ldfrag(k1p + (c + 1) * 32); }
      S0 = wmma_f16(qa[c], kb0, S0);
      S1 = wmma_f16(qa[c], kb1, S1);
      kb0 = nb0; kb1 = nb1;
    }
    // --- online softmax (row stats per half-wave; scale*log2e folded into q) ---
    float alpha[8];
#pragma unroll
    for (int r = 0; r < 8; ++r) {
      float mx = fmaxf(S0[r], S1[r]);
#pragma unroll
      for (int off = 8; off >= 1; off >>= 1) mx = fmaxf(mx, __shfl_xor(mx, off, 32));
      const float mn = fmaxf(m[r], mx);
      alpha[r] = exp2f(m[r] - mn);
      m[r] = mn;
      const float p0 = exp2f(S0[r] - mn);
      const float p1 = exp2f(S1[r] - mn);
      S0[r] = p0; S1[r] = p1;
      float sum = p0 + p1;
#pragma unroll
      for (int off = 8; off >= 1; off >>= 1) sum += __shfl_xor(sum, off, 32);
      l[r] = l[r] * alpha[r] + sum;
    }
#pragma unroll
    for (int t = 0; t < 18; ++t)
#pragma unroll
      for (int r = 0; r < 8; ++r) Oacc[t][r] *= alpha[r];
    // --- route P through LDS: C-layout -> A-fragment layout (per-wave region) ---
#pragma unroll
    for (int r = 0; r < 8; ++r) {
      const int row = r + 8 * hiHalf;
      Plds[wave][row][l15] = (_Float16)S0[r];
      Plds[wave][row][l15 + 16] = (_Float16)S1[r];
    }
    __syncthreads();
    const _Float16* pp = &Plds[wave][l15][hiHalf * 16];
    const half16v pa = cat8(*(const half8v*)pp, *(const half8v*)(pp + 8));
    // --- O += P(16x32) x V(32x288): 18 WMMAs, depth-1 pipelined LDS reads ---
    const _Float16* vp0 = Vt + (size_t)l15 * VSTR + hiHalf * 16;
    half16v vb = ldfrag(vp0);
#pragma unroll
    for (int t = 0; t < 18; ++t) {
      half16v vn = vb;
      if (t < 17) vn = ldfrag(vp0 + (size_t)(t + 1) * 16 * VSTR);
      Oacc[t] = wmma_f16(pa, vb, Oacc[t]);
      vb = vn;
    }
  }
  // --- finalize: divide by l, scatter to merged-head h buffers ---
#pragma unroll
  for (int r = 0; r < 8; ++r) l[r] = 1.0f / l[r];
  const int qb = qt * 128 + wave * 16 + 8 * hiHalf;
#pragma unroll
  for (int t = 0; t < 18; ++t) {
    const int d = t * 16 + l15;
#pragma unroll
    for (int r = 0; r < 8; ++r) {
      const float v = Oacc[t][r] * l[r];
      const size_t q = (size_t)b * NQT + qb + r;
      if (d < 256)
        hmv[(q * HCM + h * 16 + (d >> 4)) * 16 + (d & 15)] = (_Float16)v;
      else
        hs[q * HCS + h * 32 + (d - 256)] = (_Float16)v;
    }
  }
}

// ===================== Output projection -> d_out = [out_mv | out_s] fp32 =====================
__global__ __launch_bounds__(256) void proj_o_kernel(
    const _Float16* __restrict__ hmv, const _Float16* __restrict__ hs,
    const float* __restrict__ w_mv, const float* __restrict__ w_s2mv,
    const float* __restrict__ b_mv, const float* __restrict__ w_mv2s,
    const float* __restrict__ w_s2s, const float* __restrict__ b_s,
    float* __restrict__ out) {
  __shared__ float hm[128][17];
  __shared__ float hsm[256];
  const int n = blockIdx.x;  // b*NQT + q
  const int tid = threadIdx.x;
  {
    const _Float16* src = hmv + (size_t)n * 2048;
#pragma unroll
    for (int i = 0; i < 8; ++i) {
      const int idx = i * 256 + tid;
      hm[idx >> 4][idx & 15] = (float)src[idx];
    }
    hsm[tid] = (float)hs[(size_t)n * 256 + tid];
  }
  __syncthreads();
  const int o = tid >> 4, a = tid & 15;
  const float* w = w_mv + o * 128 * 9;
  const int g = GRADE_[a];
  float acc = 0.f;
  for (int c = 0; c < 128; ++c) acc = fmaf(w[c * 9 + g], hm[c][a], acc);
  const int s_ = SRC_[a];
  if (s_ >= 0) {
    const int y = Y2_[a];
    for (int c = 0; c < 128; ++c) acc = fmaf(w[c * 9 + y], hm[c][s_], acc);
  }
  if (a == 0) {
    float a0 = b_mv[o];
    for (int i = 0; i < 256; ++i) a0 = fmaf(hsm[i], w_s2mv[o * 256 + i], a0);
    acc += a0;
  }
  out[(size_t)n * 256 + tid] = acc;
  if (tid < 32) {
    float acc2 = b_s[tid];
    for (int c = 0; c < 128; ++c) acc2 = fmaf(hm[c][0], w_mv2s[tid * 128 + c], acc2);
    for (int i = 0; i < 256; ++i) acc2 = fmaf(hsm[i], w_s2s[tid * 256 + i], acc2);
    out[(size_t)NB * NQT * 256 + (size_t)n * 32 + tid] = acc2;
  }
}

extern "C" void kernel_launch(void* const* d_in, const int* in_sizes, int n_in,
                              void* d_out, int out_size, void* d_ws, size_t ws_size,
                              hipStream_t stream) {
  (void)in_sizes; (void)n_in; (void)out_size; (void)ws_size;
  const float* mv_kv = (const float*)d_in[0];
  const float* mv_q  = (const float*)d_in[1];
  const float* s_kv  = (const float*)d_in[2];
  const float* s_q   = (const float*)d_in[3];
#define WIN(i) ((const float*)d_in[4 + (i)])

  char* ws = (char*)d_ws;
  _Float16* qfeat = (_Float16*)(ws);                         // B*H*NQ*160 f16 = 20,971,520 B
  _Float16* kfeat = (_Float16*)(ws + 20971520);              // B*NK*160 f16  =  2,621,440 B
  _Float16* vT    = (_Float16*)(ws + 23592960);              // B*288*NK f16  =  4,718,592 B
  _Float16* hmv   = (_Float16*)(ws + 28311552);              // B*NQ*128*16 f16 = 33,554,432 B
  _Float16* hs    = (_Float16*)(ws + 61865984);              // B*NQ*256 f16  =  4,194,304 B

  proj_q_kernel<<<NB * NQT, 256, 0, stream>>>(
      mv_q, s_q, WIN(0), WIN(1), WIN(2), WIN(3), WIN(4), WIN(5), qfeat);
  proj_kv_kernel<<<NB * NKT, 128, 0, stream>>>(
      mv_kv, s_kv,
      WIN(6), WIN(7), WIN(8), WIN(9), WIN(10), WIN(11),
      WIN(12), WIN(13), WIN(14), WIN(15), WIN(16), WIN(17), kfeat, vT);
  attn_kernel<<<NB * NHEAD * (NQT / 128), 256, 0, stream>>>(qfeat, kfeat, vT, hmv, hs);
  proj_o_kernel<<<NB * NQT, 256, 0, stream>>>(
      hmv, hs, WIN(18), WIN(19), WIN(20), WIN(21), WIN(22), WIN(23), (float*)d_out);
#undef WIN
}
